// Joiner_70050916598490
// MI455X (gfx1250) — matrix operand order repaired and checked by reading, not compiled, not run on hardware
//
#include <hip/hip_runtime.h>
#include <hip/hip_fp16.h>
#include <math.h>

typedef __attribute__((ext_vector_type(16))) _Float16 v16h;
typedef __attribute__((ext_vector_type(8)))  _Float16 v8h;
typedef __attribute__((ext_vector_type(8)))  float    v8f;

#define N_  8
#define T_  200
#define U_  100
#define D_  512
#define V_  500
#define VPAD_ 512
#define RPB 64                   // rows per block (64 x 512 f16 act tile in LDS)
#define LDS_STRIDE (D_ + 8)      // halves; 1040 B row stride -> conflict-minimal b128 LDS ops

// Hardware tanh if this clang exposes the gfx1250 builtin; libm fallback otherwise.
#if __has_builtin(__builtin_amdgcn_tanhf)
__device__ __forceinline__ float fast_tanh(float x) { return __builtin_amdgcn_tanhf(x); }
#elif __has_builtin(__builtin_amdgcn_tanh_f32)
__device__ __forceinline__ float fast_tanh(float x) { return __builtin_amdgcn_tanh_f32(x); }
#else
__device__ __forceinline__ float fast_tanh(float x) { return tanhf(x); }
#endif

// ---------------------------------------------------------------------------
// Kernel 1: convert W (500x512 f32) -> Wh (512x512 f16, zero-padded rows)
// ---------------------------------------------------------------------------
__global__ __launch_bounds__(256) void convert_W_kernel(
    const float* __restrict__ W, _Float16* __restrict__ Wh) {
  int idx = blockIdx.x * 256 + threadIdx.x;      // over VPAD_*D_
  int v = idx / D_;
  int k = idx % D_;
  float val = (v < V_) ? W[(size_t)v * D_ + k] : 0.0f;
  Wh[idx] = (_Float16)val;
}

// ---------------------------------------------------------------------------
// Kernel 2: joiner. Block = 256 threads = 8 waves, owns 64 rows of the
// flattened (n,t,u) space. Phase 1 builds a 64x512 f16 tanh(enc+dec) tile in
// LDS; phase 2: each wave owns 4 col-tiles x 4 row-tiles, WMMA f16->f32
// against L2-resident Wh (each B fragment reused across 4 row tiles), bias,
// guarded stores for the V=500 tail.
// ---------------------------------------------------------------------------
__global__ __launch_bounds__(256) void joiner_kernel(
    const float* __restrict__ enc,   // (N,T,D)
    const float* __restrict__ dec,   // (N,U,D)
    const _Float16* __restrict__ Wh, // (VPAD_, D_) f16
    const float* __restrict__ bias,  // (V_)
    float* __restrict__ out) {       // (N*T*U, V_)

  __shared__ alignas(16) _Float16 actLds[RPB * LDS_STRIDE];

  const int block_row0 = blockIdx.x * RPB;   // global flat row base
  const int tid = threadIdx.x;

  // ---------------- Phase 1: activations -> LDS (f16) ----------------
  {
    int row_l = tid & 63;          // consecutive lanes -> consecutive rows
    int g     = tid >> 6;          // 0..3, 128 cols per thread
    int rowg  = block_row0 + row_l;
    int n   = rowg / (T_ * U_);
    int rem = rowg % (T_ * U_);
    int t = rem / U_;
    int u = rem % U_;
    const float* ep = enc + ((size_t)n * T_ + t) * D_;
    const float* dp = dec + ((size_t)n * U_ + u) * D_;
    _Float16* lp = actLds + (size_t)row_l * LDS_STRIDE;
    int cbase = g * 128;
#pragma unroll
    for (int j = 0; j < 128; j += 8) {
      int c = cbase + j;
      float4 e0 = *(const float4*)(ep + c);
      float4 e1 = *(const float4*)(ep + c + 4);
      float4 d0 = *(const float4*)(dp + c);
      float4 d1 = *(const float4*)(dp + c + 4);
      v8h h;
      h[0] = (_Float16)fast_tanh(e0.x + d0.x);
      h[1] = (_Float16)fast_tanh(e0.y + d0.y);
      h[2] = (_Float16)fast_tanh(e0.z + d0.z);
      h[3] = (_Float16)fast_tanh(e0.w + d0.w);
      h[4] = (_Float16)fast_tanh(e1.x + d1.x);
      h[5] = (_Float16)fast_tanh(e1.y + d1.y);
      h[6] = (_Float16)fast_tanh(e1.z + d1.z);
      h[7] = (_Float16)fast_tanh(e1.w + d1.w);
      *(v8h*)(lp + c) = h;           // 16B aligned
    }
  }
  __syncthreads();

  // ---------------- Phase 2: WMMA GEMM ----------------
  const int wave = tid >> 5;       // 0..7 -> col tiles [wave*4, wave*4+4)
  const int lane = tid & 31;
  const int m    = lane & 15;
  const int asel = (lane < 16) ? 0 : 16;   // A: K half-select (contiguous, matches B)
  const int bsel = (lane < 16) ? 0 : 16;   // B: K half-select (contiguous, matches A)

  const _Float16* abase = actLds + (size_t)m * LDS_STRIDE;
  const _Float16* bbase = Wh + (size_t)(wave * 4 * 16 + m) * D_;  // col = ct*16 + m

  v8f acc[4][4] = {};   // [row tile][col tile]

#pragma unroll
  for (int ks = 0; ks < D_ / 32; ++ks) {
    const int k0 = ks * 32;
    v16h afrag[4];
#pragma unroll
    for (int rt = 0; rt < 4; ++rt) {
      const _Float16* ar = abase + (size_t)rt * 16 * LDS_STRIDE;
      v8h alo = *(const v8h*)(ar + k0 + asel);
      v8h ahi = *(const v8h*)(ar + k0 + 8 + asel);
      afrag[rt] = __builtin_shufflevector(alo, ahi,
          0, 1, 2, 3, 4, 5, 6, 7, 8, 9, 10, 11, 12, 13, 14, 15);
    }
#pragma unroll
    for (int i = 0; i < 4; ++i) {
      v16h bfrag = *(const v16h*)(bbase + (size_t)i * 16 * D_ + k0 + bsel);
#pragma unroll
      for (int rt = 0; rt < 4; ++rt) {
        acc[rt][i] = __builtin_amdgcn_wmma_f32_16x16x32_f16(
            /*neg_a=*/false, afrag[rt], /*neg_b=*/false, bfrag,
            /*c_mod=*/(short)0, acc[rt][i], /*reuse_a=*/false, /*reuse_b=*/false);
      }
    }
  }

  // ---------------- Epilogue: bias + guarded store ----------------
  // C/D layout: VGPR j -> M = j (lanes 0-15) or j+8 (lanes 16-31); N = lane&15
  const int rl_off = (lane >> 4) << 3;
#pragma unroll
  for (int i = 0; i < 4; ++i) {
    int col = (wave * 4 + i) * 16 + m;
    if (col < V_) {
      float bv = bias[col];
      float* ocol = out + (size_t)block_row0 * V_ + col;
#pragma unroll
      for (int rt = 0; rt < 4; ++rt) {
#pragma unroll
        for (int j = 0; j < 8; ++j) {
          int row_local = rt * 16 + rl_off + j;
          ocol[(size_t)row_local * V_] = acc[rt][i][j] + bv;
        }
      }
    }
  }
}

// ---------------------------------------------------------------------------
// Launch
// ---------------------------------------------------------------------------
extern "C" void kernel_launch(void* const* d_in, const int* in_sizes, int n_in,
                              void* d_out, int out_size, void* d_ws, size_t ws_size,
                              hipStream_t stream) {
  const float* enc = (const float*)d_in[0];   // (8,200,512)
  const float* dec = (const float*)d_in[1];   // (8,100,512)
  const float* W   = (const float*)d_in[2];   // (500,512)
  const float* b   = (const float*)d_in[3];   // (500,)
  float* out = (float*)d_out;                 // (8,200,100,500)

  _Float16* Wh = (_Float16*)d_ws;             // needs VPAD_*D_*2 = 512 KB scratch

  convert_W_kernel<<<(VPAD_ * D_) / 256, 256, 0, stream>>>(W, Wh);

  const int total_rows = N_ * T_ * U_;        // 160000, divisible by 64
  joiner_kernel<<<total_rows / RPB, 256, 0, stream>>>(enc, dec, Wh, b, out);
}
